// PiecewiseRationalQuadraticCoupling_6811818132468
// MI455X (gfx1250) — matrix operand order repaired and checked by reading, not compiled, not run on hardware
//
#include <hip/hip_runtime.h>
#include <hip/hip_bf16.h>
#include <math.h>

typedef __attribute__((ext_vector_type(16))) _Float16 v16h;
typedef __attribute__((ext_vector_type(8)))  _Float16 v8h;
typedef __attribute__((ext_vector_type(8)))  float    v8f;

#define HIDN   256
#define NOUTC  800     // D_TR * (3K+1) = 32*25
#define NPARAM 25
#define M_BLK  32      // rows per workgroup (2 M-tiles -> B-fragment reuse x2)
#define HSTR   264     // padded hidden stride (halves): 528B row stride, conflict-free b128
#define PSTR   804     // padded params stride (floats): 25*d mod 64 banks all distinct

// ---------------- weight pre-pack kernels (B-fragment layout, f16) -------------
// B fragment (f16, KxN = 32x16): lane L holds column N = L&15, K-half hs = L>>4,
// element e -> K = (e&7) + 8*hs + 16*(e>>3).  Packed as [tile][lane][e] so each lane reads one
// contiguous 32-byte v16h.

__global__ void pack_w1(const float* __restrict__ W1, _Float16* __restrict__ out) {
    int idx = blockIdx.x * blockDim.x + threadIdx.x;
    if (idx >= 16 * 512) return;              // 16 n-tiles * 32 lanes * 16 halves
    int ntile = idx >> 9;
    int rem   = idx & 511;
    int lane  = rem >> 4;
    int e     = rem & 15;
    int k     = (e & 7) + ((lane >> 4) << 3) + ((e >> 3) << 4);       // 0..31
    int n     = ntile * 16 + (lane & 15);
    out[idx]  = (_Float16)W1[k * HIDN + n];
}

__global__ void pack_w2(const float* __restrict__ W2, _Float16* __restrict__ out) {
    int idx = blockIdx.x * blockDim.x + threadIdx.x;
    if (idx >= 400 * 512) return;             // (50 n-tiles * 8 k-steps) * 32 * 16
    int blk  = idx >> 9;                      // n*8 + ks
    int rem  = idx & 511;
    int lane = rem >> 4;
    int e    = rem & 15;
    int ks   = blk & 7;
    int n    = blk >> 3;
    int k    = ks * 32 + ((e >> 3) << 4) + ((lane >> 4) << 3) + (e & 7);
    int col  = n * 16 + (lane & 15);
    out[idx] = (_Float16)W2[k * NOUTC + col];
}

// ---------------- fused MLP + rational-quadratic-spline kernel -----------------

__global__ void __launch_bounds__(256)
rqs_coupling_fused(const float* __restrict__ inp,
                   const float* __restrict__ b1,
                   const float* __restrict__ b2,
                   const _Float16* __restrict__ w1p,
                   const _Float16* __restrict__ w2p,
                   float* __restrict__ out,
                   int nRows)
{
    extern __shared__ char smem[];
    _Float16* sh_h   = (_Float16*)smem;                                  // 32*264 halves
    float*    sh_p   = (float*)(smem + M_BLK * HSTR * 2);                // 32*804 floats
    float*    sh_lad = (float*)(smem + M_BLK * HSTR * 2 + M_BLK * PSTR * 4);

    const int  tid  = threadIdx.x;
    const int  wave = tid >> 5;
    const int  lane = tid & 31;
    const int  nl   = lane & 15;
    const int  hs   = lane >> 4;
    const long rowBase = (long)blockIdx.x * M_BLK;

    if (tid < M_BLK) sh_lad[tid] = 0.0f;

    // -------- GEMM1: hidden = relu(identity @ W1 + b1), one K=32 WMMA/tile -----
    // A fragment (f16 16x32): lane L: row M = L&15, e<8 -> K=8*hs+e, e>=8 -> K=16+8*hs+(e-8)
    v16h a0, a1;
    {
        const float* rp0 = inp + (rowBase + nl) * 64;
        const float* rp1 = inp + (rowBase + 16 + nl) * 64;
        #pragma unroll
        for (int m = 0; m < 2; ++m) {
            const float* rp = m ? rp1 : rp0;
            float4 f0 = *(const float4*)(rp + 8 * hs);
            float4 f1 = *(const float4*)(rp + 8 * hs + 4);
            float4 f2 = *(const float4*)(rp + 16 + 8 * hs);
            float4 f3 = *(const float4*)(rp + 20 + 8 * hs);
            v16h a;
            a[0]  = (_Float16)f0.x; a[1]  = (_Float16)f0.y;
            a[2]  = (_Float16)f0.z; a[3]  = (_Float16)f0.w;
            a[4]  = (_Float16)f1.x; a[5]  = (_Float16)f1.y;
            a[6]  = (_Float16)f1.z; a[7]  = (_Float16)f1.w;
            a[8]  = (_Float16)f2.x; a[9]  = (_Float16)f2.y;
            a[10] = (_Float16)f2.z; a[11] = (_Float16)f2.w;
            a[12] = (_Float16)f3.x; a[13] = (_Float16)f3.y;
            a[14] = (_Float16)f3.z; a[15] = (_Float16)f3.w;
            if (m) a1 = a; else a0 = a;
        }
    }
    for (int n = wave; n < 16; n += 8) {
        v16h bf   = *((const v16h*)w1p + (n * 32 + lane));
        float bias = b1[n * 16 + nl];
        v8f acc0, acc1;
        #pragma unroll
        for (int r = 0; r < 8; ++r) { acc0[r] = bias; acc1[r] = bias; }
        acc0 = __builtin_amdgcn_wmma_f32_16x16x32_f16(false, a0, false, bf,
                                                      (short)0, acc0, false, false);
        acc1 = __builtin_amdgcn_wmma_f32_16x16x32_f16(false, a1, false, bf,
                                                      (short)0, acc1, false, false);
        // C layout: VGPR r -> row M = r + 8*hs, col N = nl
        #pragma unroll
        for (int r = 0; r < 8; ++r) {
            sh_h[(r + 8 * hs) * HSTR + n * 16 + nl]        = (_Float16)fmaxf(acc0[r], 0.0f);
            sh_h[(16 + r + 8 * hs) * HSTR + n * 16 + nl]   = (_Float16)fmaxf(acc1[r], 0.0f);
        }
    }
    __syncthreads();

    // -------- GEMM2: params = hidden @ W2 + b2  (K=256 -> 8 WMMA steps) --------
    // Hoist both M-tiles' A fragments; each 32B B fragment feeds 2 WMMAs.
    v16h af0[8], af1[8];
    #pragma unroll
    for (int ks = 0; ks < 8; ++ks) {
        const _Float16* hp0 = sh_h + nl * HSTR + ks * 32 + 8 * hs;
        const _Float16* hp1 = hp0 + 16 * HSTR;
        v8h lo0 = *(const v8h*)hp0;
        v8h hi0 = *(const v8h*)(hp0 + 16);
        v8h lo1 = *(const v8h*)hp1;
        v8h hi1 = *(const v8h*)(hp1 + 16);
        #pragma unroll
        for (int i = 0; i < 8; ++i) {
            af0[ks][i] = lo0[i]; af0[ks][8 + i] = hi0[i];
            af1[ks][i] = lo1[i]; af1[ks][8 + i] = hi1[i];
        }
    }
    for (int n = wave; n < 50; n += 8) {
        float bias = b2[n * 16 + nl];
        v8f acc0, acc1;
        #pragma unroll
        for (int r = 0; r < 8; ++r) { acc0[r] = bias; acc1[r] = bias; }
        const v16h* bp = (const v16h*)w2p + (size_t)(n * 8) * 32 + lane;
        // prefetch next n-tile's B fragments for this wave (L2-resident stream)
        __builtin_prefetch((const void*)(bp + 8 * 8 * 32), 0, 1);
        #pragma unroll
        for (int ks = 0; ks < 8; ++ks) {
            v16h bf = bp[ks * 32];
            acc0 = __builtin_amdgcn_wmma_f32_16x16x32_f16(false, af0[ks], false, bf,
                                                          (short)0, acc0, false, false);
            acc1 = __builtin_amdgcn_wmma_f32_16x16x32_f16(false, af1[ks], false, bf,
                                                          (short)0, acc1, false, false);
        }
        #pragma unroll
        for (int r = 0; r < 8; ++r) {
            sh_p[(r + 8 * hs) * PSTR + n * 16 + nl]      = acc0[r];
            sh_p[(16 + r + 8 * hs) * PSTR + n * 16 + nl] = acc1[r];
        }
    }
    __syncthreads();

    // -------- identity passthrough (float4 vectorized) --------
    for (int i = tid; i < M_BLK * 8; i += 256) {
        int r = i >> 3, c4 = i & 7;
        *(float4*)(out + (rowBase + r) * 64 + c4 * 4) =
            *(const float4*)(inp + (rowBase + r) * 64 + c4 * 4);
    }

    // -------- rational quadratic spline per (row, dim) --------
    for (int i = tid; i < M_BLK * 32; i += 256) {
        int row = i >> 5, d = i & 31;
        const float* pp = sh_p + row * PSTR + d * NPARAM;

        float uw[8], uh[8], ud[9];
        #pragma unroll
        for (int j = 0; j < 8; ++j) { uw[j] = pp[j]; uh[j] = pp[8 + j]; }
        #pragma unroll
        for (int j = 0; j < 9; ++j) ud[j] = pp[16 + j];

        // widths: softmax -> affine -> cumsum -> renormalize
        float mw = uw[0];
        #pragma unroll
        for (int j = 1; j < 8; ++j) mw = fmaxf(mw, uw[j]);
        float ew[8], sw = 0.0f;
        #pragma unroll
        for (int j = 0; j < 8; ++j) { ew[j] = expf(uw[j] - mw); sw += ew[j]; }
        float cw[9]; cw[0] = 0.0f;
        {
            float run = 0.0f, invs = 1.0f / sw;
            #pragma unroll
            for (int j = 0; j < 8; ++j) { run += 0.001f + 0.992f * (ew[j] * invs); cw[j + 1] = run; }
            float inv = 1.0f / fmaxf(cw[8], 1e-12f);
            #pragma unroll
            for (int j = 0; j < 9; ++j) cw[j] *= inv;
        }
        // heights
        float mh = uh[0];
        #pragma unroll
        for (int j = 1; j < 8; ++j) mh = fmaxf(mh, uh[j]);
        float eh[8], shh = 0.0f;
        #pragma unroll
        for (int j = 0; j < 8; ++j) { eh[j] = expf(uh[j] - mh); shh += eh[j]; }
        float ch[9]; ch[0] = 0.0f;
        {
            float run = 0.0f, invs = 1.0f / shh;
            #pragma unroll
            for (int j = 0; j < 8; ++j) { run += 0.001f + 0.992f * (eh[j] * invs); ch[j + 1] = run; }
            float inv = 1.0f / fmaxf(ch[8], 1e-12f);
            #pragma unroll
            for (int j = 0; j < 9; ++j) ch[j] *= inv;
        }
        // derivatives: MIN_D + softplus
        float dv[9];
        #pragma unroll
        for (int j = 0; j < 9; ++j) {
            float u  = ud[j];
            float sp = (u > 20.0f) ? u : log1pf(expf(u));
            dv[j] = 0.001f + sp;
        }

        float x = inp[(rowBase + row) * 64 + 32 + d];
        bool inside = (x >= -10.0f) && (x <= 10.0f);
        float xs = fminf(fmaxf((x + 10.0f) * 0.05f, 0.0f), 1.0f);

        int cnt = 0;
        #pragma unroll
        for (int j = 0; j < 9; ++j) cnt += (xs >= cw[j]) ? 1 : 0;
        int bin = cnt - 1;
        if (bin < 0) bin = 0;
        if (bin > 7) bin = 7;

        float xk = cw[bin], wk = cw[bin + 1] - cw[bin];
        float yk = ch[bin], hk = ch[bin + 1] - ch[bin];
        float dk = dv[bin], dk1 = dv[bin + 1];

        float t   = fminf(fmaxf((xs - xk) / (wk + 1e-12f), 0.0f), 1.0f);
        float a   = (hk + 1e-12f) / (wk + 1e-12f);
        float omt = 1.0f - t;
        float num = a * t * t + dk * t * omt;
        float den = a + (dk + dk1 - 2.0f * a) * t * omt;
        float s   = num / (den + 1e-12f);
        float y   = (yk + hk * s) * 20.0f - 10.0f;
        float dn  = a * a * (dk1 * t * t + 2.0f * a * t * omt + dk * omt * omt);
        float dydx = dn / (den * den + 1e-12f);
        float lad  = logf(fmaxf(dydx, 1e-12f));

        out[(rowBase + row) * 64 + 32 + d] = inside ? y : x;
        atomicAdd(&sh_lad[row], inside ? lad : 0.0f);
    }
    __syncthreads();

    if (tid < M_BLK)
        out[(long)nRows * 64 + rowBase + tid] = sh_lad[tid];
}

// ------------------------------- launcher --------------------------------------

extern "C" void kernel_launch(void* const* d_in, const int* in_sizes, int n_in,
                              void* d_out, int out_size, void* d_ws, size_t ws_size,
                              hipStream_t stream) {
    const float* inp = (const float*)d_in[0];   // (B, 64) f32
    const float* W1  = (const float*)d_in[1];   // (32, 256)
    const float* b1  = (const float*)d_in[2];   // (256,)
    const float* W2  = (const float*)d_in[3];   // (256, 800)
    const float* b2  = (const float*)d_in[4];   // (800,)
    float* out = (float*)d_out;                 // B*64 outputs then B logabsdet

    int nRows = in_sizes[0] / 64;

    _Float16* w1p = (_Float16*)d_ws;                        // 16 KB
    _Float16* w2p = (_Float16*)((char*)d_ws + 16384);       // 400 KB

    pack_w1<<<(16 * 512 + 255) / 256, 256, 0, stream>>>(W1, w1p);
    pack_w2<<<(400 * 512 + 255) / 256, 256, 0, stream>>>(W2, w2p);

    size_t smem = (size_t)M_BLK * HSTR * 2 + (size_t)M_BLK * PSTR * 4 + M_BLK * 4;
    int grid = nRows / M_BLK;
    rqs_coupling_fused<<<grid, 256, smem, stream>>>(inp, b1, b2, w1p, w2p, out, nRows);
}
